// VectorQuantizer_55379308315292
// MI455X (gfx1250) — matrix operand order repaired and checked by reading, not compiled, not run on hardware
//
#include <hip/hip_runtime.h>
#include <hip/hip_bf16.h>

typedef __attribute__((ext_vector_type(16))) __bf16 v16bf;
typedef __attribute__((ext_vector_type(8)))  __bf16 v8bf;
typedef __attribute__((ext_vector_type(8)))  float  v8f;

#define N_E   1024
#define E_DIM 2560
#define B_SZ  2048
#define SPP   32768      // 128*16*16 floats per structure row
#define BETA  0.25f

// ---------------------------------------------------------------- init winner
__global__ void vq_init(int* __restrict__ winner) {
    int i = blockIdx.x * blockDim.x + threadIdx.x;
    if (i < N_E) winner[i] = -1;
}

// ------------------------------------------------ f32 -> bf16 hi/lo splitting
__global__ void vq_split(const float* __restrict__ src,
                         __bf16* __restrict__ hi, __bf16* __restrict__ lo, int n) {
    int i = blockIdx.x * blockDim.x + threadIdx.x;
    int stride = gridDim.x * blockDim.x;
    for (; i < n; i += stride) {
        float f = src[i];
        __bf16 h = (__bf16)f;
        hi[i] = h;
        lo[i] = (__bf16)(f - (float)h);
    }
}

// ---------------------------------------------------------------- ||e_n||^2
__global__ void vq_enorm(const float* __restrict__ emb, float* __restrict__ enorm) {
    __shared__ float red[256];
    const float* p = emb + (long)blockIdx.x * E_DIM;
    float s = 0.f;
    for (int k = threadIdx.x; k < E_DIM; k += 256) { float v = p[k]; s += v * v; }
    red[threadIdx.x] = s; __syncthreads();
    for (int off = 128; off; off >>= 1) {
        if (threadIdx.x < off) red[threadIdx.x] += red[threadIdx.x + off];
        __syncthreads();
    }
    if (threadIdx.x == 0) enorm[blockIdx.x] = red[0];
}

// --------------------------------------------------- WMMA distance + argmin
// Grid: 128 blocks (one 16-row tile of z each), 128 threads = 4 waves.
// Each wave owns 16 of the 64 column tiles, accumulating 4 tiles at a time so
// the A fragment is reused across 4 independent WMMA chains.
__global__ __launch_bounds__(128) void vq_argmin(
        const __bf16* __restrict__ zhi, const __bf16* __restrict__ zlo,
        const __bf16* __restrict__ ehi, const __bf16* __restrict__ elo,
        const float* __restrict__ enorm,
        int* __restrict__ idx_out, int* __restrict__ winner) {
    __shared__ float sval[4][16];
    __shared__ int   sidx[4][16];

    const int wave = threadIdx.x >> 5;
    const int lane = threadIdx.x & 31;
    const int rit  = lane & 15;      // row-in-tile / col-in-tile
    const int half = lane >> 4;      // lane half selects K sub-range
    const int m0   = blockIdx.x * 16;

    const __bf16* za = zhi + (long)(m0 + rit) * E_DIM;
    const __bf16* zb = zlo + (long)(m0 + rit) * E_DIM;

    float best[8]; int bidx[8];
    #pragma unroll
    for (int r = 0; r < 8; ++r) { best[r] = 3.4e38f; bidx[r] = 0x7fffffff; }

    for (int gg = 0; gg < 4; ++gg) {
        const int t0 = (wave + 4 * gg) * 4;          // 4 consecutive N tiles
        v8f acc[4];
        #pragma unroll
        for (int j = 0; j < 4; ++j) acc[j] = v8f{0.f,0.f,0.f,0.f,0.f,0.f,0.f,0.f};

        for (int kk = 0; kk < E_DIM; kk += 32) {
            // A fragment: 16x32 bf16 (ISA layout: two 8-elem K chunks per lane)
            const int ka = kk + half * 8;
            v8bf a0h = *(const v8bf*)(za + ka);
            v8bf a1h = *(const v8bf*)(za + ka + 16);
            v8bf a0l = *(const v8bf*)(zb + ka);
            v8bf a1l = *(const v8bf*)(zb + ka + 16);
            v16bf ahi = __builtin_shufflevector(a0h, a1h, 0,1,2,3,4,5,6,7,8,9,10,11,12,13,14,15);
            v16bf alo = __builtin_shufflevector(a0l, a1l, 0,1,2,3,4,5,6,7,8,9,10,11,12,13,14,15);
            #pragma unroll
            for (int j = 0; j < 4; ++j) {
                const int col = (t0 + j) * 16 + rit;  // B column == emb row
                const __bf16* pbh = ehi + (long)col * E_DIM + kk + half * 8;
                const __bf16* pbl = elo + (long)col * E_DIM + kk + half * 8;
                v16bf bhi = __builtin_shufflevector(*(const v8bf*)pbh, *(const v8bf*)(pbh + 16), 0,1,2,3,4,5,6,7,8,9,10,11,12,13,14,15);
                v16bf blo = __builtin_shufflevector(*(const v8bf*)pbl, *(const v8bf*)(pbl + 16), 0,1,2,3,4,5,6,7,8,9,10,11,12,13,14,15);
                // hi*hi + hi*lo + lo*hi  ~ f32-accurate dot (lo*lo negligible)
                acc[j] = __builtin_amdgcn_wmma_f32_16x16x32_bf16(false, ahi, false, bhi, (short)0, acc[j], false, false);
                acc[j] = __builtin_amdgcn_wmma_f32_16x16x32_bf16(false, ahi, false, blo, (short)0, acc[j], false, false);
                acc[j] = __builtin_amdgcn_wmma_f32_16x16x32_bf16(false, alo, false, bhi, (short)0, acc[j], false, false);
            }
        }
        // d = ||e_n||^2 - 2 z.e   (||z||^2 constant per row, dropped)
        #pragma unroll
        for (int j = 0; j < 4; ++j) {
            const int n = (t0 + j) * 16 + rit;
            const float en = enorm[n];
            #pragma unroll
            for (int r = 0; r < 8; ++r) {
                float dv = en - 2.f * acc[j][r];
                bool better = (dv < best[r]) | ((dv == best[r]) & (n < bidx[r]));
                best[r] = better ? dv : best[r];
                bidx[r] = better ? n  : bidx[r];
            }
        }
    }

    // Cross-lane argmin within each 16-lane half (C layout: lanes 0-15 -> M=r,
    // lanes 16-31 -> M=r+8, N = lane%16).
    #pragma unroll
    for (int r = 0; r < 8; ++r) {
        #pragma unroll
        for (int off = 8; off; off >>= 1) {
            float ov = __shfl_xor(best[r], off, 16);
            int   oi = __shfl_xor(bidx[r], off, 16);
            bool better = (ov < best[r]) | ((ov == best[r]) & (oi < bidx[r]));
            best[r] = better ? ov : best[r];
            bidx[r] = better ? oi : bidx[r];
        }
    }
    if (rit == 0) {
        #pragma unroll
        for (int r = 0; r < 8; ++r) {
            sval[wave][r + half * 8] = best[r];
            sidx[wave][r + half * 8] = bidx[r];
        }
    }
    __syncthreads();
    if (threadIdx.x < 16) {
        float bv = sval[0][threadIdx.x]; int bi = sidx[0][threadIdx.x];
        #pragma unroll
        for (int w = 1; w < 4; ++w) {
            float ov = sval[w][threadIdx.x]; int oi = sidx[w][threadIdx.x];
            bool better = (ov < bv) | ((ov == bv) & (oi < bi));
            bv = better ? ov : bv; bi = better ? oi : bi;
        }
        const int row = m0 + threadIdx.x;
        idx_out[row] = bi;
        atomicMax(&winner[bi], row);   // order-independent "last write wins"
    }
}

// ------------------------------------- z_q gather, straight-through, loss part
__global__ void vq_gather(const float* __restrict__ z, const float* __restrict__ emb,
                          const int* __restrict__ idx, float* __restrict__ zq_out,
                          float* __restrict__ partials) {
    __shared__ float red[256];
    const int b = blockIdx.x;
    const float* e  = emb + (long)idx[b] * E_DIM;
    const float* zr = z   + (long)b * E_DIM;
    float*       o  = zq_out + (long)b * E_DIM;
    float s = 0.f;
    for (int k = threadIdx.x; k < E_DIM; k += 256) {
        float zq = e[k], zv = zr[k];
        float d = zq - zv;
        o[k] = zv + d;        // z + sg(z_q - z), bitwise like reference
        s += d * d;
    }
    red[threadIdx.x] = s; __syncthreads();
    for (int off = 128; off; off >>= 1) {
        if (threadIdx.x < off) red[threadIdx.x] += red[threadIdx.x + off];
        __syncthreads();
    }
    if (threadIdx.x == 0) partials[b] = red[0];
}

__global__ void vq_loss(const float* __restrict__ partials, float* __restrict__ loss) {
    __shared__ float red[256];
    float s = 0.f;
    for (int i = threadIdx.x; i < B_SZ; i += 256) s += partials[i];
    red[threadIdx.x] = s; __syncthreads();
    for (int off = 128; off; off >>= 1) {
        if (threadIdx.x < off) red[threadIdx.x] += red[threadIdx.x + off];
        __syncthreads();
    }
    if (threadIdx.x == 0)
        loss[0] = (1.f + BETA) * red[0] / (float)((long)B_SZ * E_DIM);
}

// --------------------------- single streaming pass producing the merged bank
__global__ void vq_bank(const float* __restrict__ bank, const float* __restrict__ structures,
                        const int* __restrict__ winner, float* __restrict__ out) {
    const int n = blockIdx.x;
    const int w = winner[n];
    const float4* src = (const float4*)(w >= 0 ? structures + (long)w * SPP
                                               : bank       + (long)n * SPP);
    float4* dst = (float4*)(out + (long)n * SPP);
    for (int i = threadIdx.x; i < SPP / 4; i += blockDim.x) dst[i] = src[i];
}

// ----------------------------------------------------------------- launcher
extern "C" void kernel_launch(void* const* d_in, const int* in_sizes, int n_in,
                              void* d_out, int out_size, void* d_ws, size_t ws_size,
                              hipStream_t stream) {
    const float* z          = (const float*)d_in[0];   // [2048, 2560]
    const float* structures = (const float*)d_in[1];   // [2048, 128,16,16]
    const float* emb        = (const float*)d_in[2];   // [1024, 2560]
    const float* bank       = (const float*)d_in[3];   // [1024, 128,16,16]

    float* out   = (float*)d_out;
    float* loss  = out;                                 // [1]
    float* zq    = out + 1;                             // [2048*2560]
    float* obank = out + 1 + (long)B_SZ * E_DIM;        // [1024*32768]

    char* ws = (char*)d_ws;
    int*    idx      = (int*)ws;                        // 2048 ints
    int*    winner   = (int*)(ws + 8192);               // 1024 ints
    float*  partials = (float*)(ws + 12288);            // 2048 floats
    float*  enorm    = (float*)(ws + 20480);            // 1024 floats
    __bf16* zhi      = (__bf16*)(ws + 24576);           // 64B-aligned
    __bf16* zlo      = zhi + (size_t)B_SZ * E_DIM;
    __bf16* ehi      = zlo + (size_t)B_SZ * E_DIM;
    __bf16* elo      = ehi + (size_t)N_E * E_DIM;       // total ~31.5 MB

    vq_init <<<4, 256, 0, stream>>>(winner);
    vq_split<<<2048, 256, 0, stream>>>(z,   zhi, zlo, B_SZ * E_DIM);
    vq_split<<<1024, 256, 0, stream>>>(emb, ehi, elo, N_E * E_DIM);
    vq_enorm<<<N_E, 256, 0, stream>>>(emb, enorm);
    vq_argmin<<<B_SZ / 16, 128, 0, stream>>>(zhi, zlo, ehi, elo, enorm, idx, winner);
    vq_gather<<<B_SZ, 256, 0, stream>>>(z, emb, idx, zq, partials);
    vq_loss  <<<1, 256, 0, stream>>>(partials, loss);
    vq_bank  <<<N_E, 256, 0, stream>>>(bank, structures, winner, obank);
}
